// MessagePassingNetwork_37855841747621
// MI455X (gfx1250) — matrix operand order repaired and checked by reading, not compiled, not run on hardware
//
#include <hip/hip_runtime.h>
#include <hip/hip_bf16.h>
#include <stdint.h>

// ---------------------------------------------------------------------------
// MessagePassingNetwork on gfx1250: bf16 WMMA tiles, f32 accumulate.
// Weights pre-swizzled into per-lane fragment order so B fragments load as
// two ds_load_b128 per WMMA operand (no cross-register shuffles).
// ---------------------------------------------------------------------------

typedef __bf16 bf16;
typedef bf16  v16bf __attribute__((ext_vector_type(16)));
typedef float v8f   __attribute__((ext_vector_type(8)));

#define N_NODES 100000
#define N_EDGES 1000000
#define HID 64
#define ETILES (N_EDGES / 16)   // 62500
#define NTILES (N_NODES / 16)   // 6250
#define TPW 8                   // tiles per wave

// ---- WMMA wrapper ----------------------------------------------------------
__device__ __forceinline__ v8f wmma_bf16(v16bf a, v16bf b, v8f c) {
  return __builtin_amdgcn_wmma_f32_16x16x32_bf16(false, a, false, b, (short)0,
                                                 c, false, false);
}

// Wave-local LDS fence: all prior DS ops complete (per-wave tiles only).
__device__ __forceinline__ void wave_lds_fence() {
  asm volatile("s_wait_dscnt 0" ::: "memory");
}

__device__ __forceinline__ uint32_t pack2bf(float x, float y) {
  union { bf16 h[2]; uint32_t u; } t;
  t.h[0] = (bf16)x;
  t.h[1] = (bf16)y;
  return t.u;
}

// ---- A fragment: 16x32 bf16 tile (row-major in LDS, stride in elements) ----
// lanes 0-15: M=lane, K {0..15} in v0-7; lanes 16-31: K {16..31}.
__device__ __forceinline__ v16bf load_a_frag(const bf16* tile, int stride,
                                             int kb, int lane) {
  const int m = lane & 15;
  const int h = (lane >> 4) << 3;  // dword offset 0 or 8
  v16bf a;
  uint32_t* ap = reinterpret_cast<uint32_t*>(&a);
  const uint32_t* base =
      reinterpret_cast<const uint32_t*>(tile + m * stride + kb);
#pragma unroll
  for (int v = 0; v < 4; ++v) ap[v] = base[h + v];
#pragma unroll
  for (int v = 0; v < 4; ++v) ap[4 + v] = base[4 + h + v];
  return a;
}

// ---- B fragment from swizzled weights: contiguous 8 dwords per lane --------
__device__ __forceinline__ v16bf load_b_frag(const bf16* sw, int frag,
                                             int lane) {
  const uint32_t* p =
      reinterpret_cast<const uint32_t*>(sw) + ((frag * 32 + lane) << 3);
  v16bf b;
  uint32_t* bp = reinterpret_cast<uint32_t*>(&b);
#pragma unroll
  for (int j = 0; j < 8; ++j) bp[j] = p[j];
  return b;
}

__device__ __forceinline__ float swish(float x) {
  return x / (1.0f + __expf(-x));
}

// ---- weight pack: f32 row-major (K x 64) -> fragment-swizzled bf16 ---------
// dword d = ((ks*4+nb)*32 + lane)*8 + j  holds bf16 pair (k, k+1) at col n
//   n = (lane&15) + nb*16 ;  k = ks*32 + ((lane>>4)<<4) + 2*j
__global__ void pack_w_kernel(const float* __restrict__ w,
                              uint32_t* __restrict__ o, int totalDwords) {
  int d = blockIdx.x * blockDim.x + threadIdx.x;
  if (d >= totalDwords) return;
  const int f = d >> 8, rem = d & 255, lane = rem >> 3, j = rem & 7;
  const int ks = f >> 2, nb = f & 3;
  const int n = (lane & 15) + nb * 16;
  const int k = ks * 32 + ((lane >> 4) << 4) + 2 * j;
  o[d] = pack2bf(w[k * 64 + n], w[(k + 1) * 64 + n]);
}

__global__ void zero_kernel(float* __restrict__ p, int n) {
  int i = blockIdx.x * blockDim.x + threadIdx.x;
  if (i < n) p[i] = 0.0f;
}

// ---------------------------------------------------------------------------
// Edge / message kernel. 2 waves/block, TPW 16-edge tiles per wave.
//   h = [nodes[s] | nodes[r] | edges]  (16 x 192)
//   msg = swish(h @ W1 + b1) @ W2 + b2 ; agg[recv] += msg  (atomic f32)
// ---------------------------------------------------------------------------
__global__ __launch_bounds__(64) void edge_kernel(
    const float* __restrict__ nodes, const float* __restrict__ edges,
    const int* __restrict__ senders, const int* __restrict__ receivers,
    const bf16* __restrict__ w1sw, const bf16* __restrict__ w2sw,
    const float* __restrict__ b1, const float* __restrict__ b2,
    float* __restrict__ agg) {
  __shared__ alignas(16) bf16 sW1[192 * 64];     // 24 KB (24 frags)
  __shared__ alignas(16) bf16 sW2[64 * 64];      //  8 KB (8 frags)
  __shared__ alignas(16) bf16 sA1[2][16 * 192];  // 12 KB
  __shared__ alignas(16) bf16 sA2[2][16 * 64];   //  4 KB

  const int tid  = threadIdx.x;
  const int lane = tid & 31;
  const int w    = tid >> 5;

  {  // stage weights once per block (uint4 copies)
    uint4* d1 = reinterpret_cast<uint4*>(sW1);
    const uint4* s1 = reinterpret_cast<const uint4*>(w1sw);
    for (int i = tid; i < 192 * 64 / 8; i += 64) d1[i] = s1[i];
    uint4* d2 = reinterpret_cast<uint4*>(sW2);
    const uint4* s2 = reinterpret_cast<const uint4*>(w2sw);
    for (int i = tid; i < 64 * 64 / 8; i += 64) d2[i] = s2[i];
  }
  __syncthreads();

  const int gw = blockIdx.x * 2 + w;
  bf16* A1 = sA1[w];
  bf16* A2 = sA2[w];
  const int cn = lane & 15;
  const int rb = (lane >> 4) << 3;

  for (int it = 0; it < TPW; ++it) {
    const int tile = gw * TPW + it;   // wave-uniform
    if (tile >= ETILES) break;
    const int e0 = tile * 16;

    // Stage A1 tile: float4 gathers, packed bf16, ds_store_b64.
    for (int idx = lane; idx < 16 * 16; idx += 32) {
      const int er = idx >> 4, c4 = (idx & 15) << 2;
      const int s  = senders[e0 + er];
      const int rv = receivers[e0 + er];
      float4 fa = *(const float4*)&nodes[s * 64 + c4];
      float4 fb = *(const float4*)&nodes[rv * 64 + c4];
      float4 fc = *(const float4*)&edges[(e0 + er) * 64 + c4];
      uint32_t* drow = reinterpret_cast<uint32_t*>(&A1[er * 192]);
      *(uint2*)&drow[c4 >> 1] =
          make_uint2(pack2bf(fa.x, fa.y), pack2bf(fa.z, fa.w));
      *(uint2*)&drow[(64 + c4) >> 1] =
          make_uint2(pack2bf(fb.x, fb.y), pack2bf(fb.z, fb.w));
      *(uint2*)&drow[(128 + c4) >> 1] =
          make_uint2(pack2bf(fc.x, fc.y), pack2bf(fc.z, fc.w));
    }
    wave_lds_fence();

    // GEMM1: (16x192) @ (192x64)
    v8f acc[4] = {};
#pragma unroll
    for (int ks = 0; ks < 6; ++ks) {
      v16bf a = load_a_frag(A1, 192, ks * 32, lane);
#pragma unroll
      for (int nb = 0; nb < 4; ++nb) {
        v16bf bm = load_b_frag(sW1, ks * 4 + nb, lane);
        acc[nb] = wmma_bf16(a, bm, acc[nb]);
      }
    }

    // bias + swish -> A2 (bf16)
#pragma unroll
    for (int nb = 0; nb < 4; ++nb) {
      const float bias = b1[nb * 16 + cn];
#pragma unroll
      for (int r = 0; r < 8; ++r) {
        float x = acc[nb][r] + bias;
        A2[(rb + r) * 64 + nb * 16 + cn] = (bf16)swish(x);
      }
    }
    wave_lds_fence();

    // GEMM2: (16x64) @ (64x64)
    v8f acc2[4] = {};
#pragma unroll
    for (int ks = 0; ks < 2; ++ks) {
      v16bf a = load_a_frag(A2, 64, ks * 32, lane);
#pragma unroll
      for (int nb = 0; nb < 4; ++nb) {
        v16bf bm = load_b_frag(sW2, ks * 4 + nb, lane);
        acc2[nb] = wmma_bf16(a, bm, acc2[nb]);
      }
    }

    // bias + atomic segment-sum scatter (receivers sorted -> L2 locality)
#pragma unroll
    for (int nb = 0; nb < 4; ++nb) {
      const float bias = b2[nb * 16 + cn];
#pragma unroll
      for (int r = 0; r < 8; ++r) {
        const int rv = receivers[e0 + rb + r];
        atomicAdd(&agg[rv * 64 + nb * 16 + cn], acc2[nb][r] + bias);
      }
    }
    wave_lds_fence();  // A1/A2 reused next iteration
  }
}

// ---------------------------------------------------------------------------
// Node update kernel: nodes' = swish([nodes|agg] @ uW1 + ub1) @ uW2 + ub2
// ---------------------------------------------------------------------------
__global__ __launch_bounds__(64) void update_kernel(
    const float* __restrict__ nodes, const float* __restrict__ agg,
    const bf16* __restrict__ w1sw, const bf16* __restrict__ w2sw,
    const float* __restrict__ b1, const float* __restrict__ b2,
    float* __restrict__ out) {
  __shared__ alignas(16) bf16 sW1[128 * 64];     // 16 KB (16 frags)
  __shared__ alignas(16) bf16 sW2[64 * 64];      //  8 KB
  __shared__ alignas(16) bf16 sA1[2][16 * 128];  //  8 KB
  __shared__ alignas(16) bf16 sA2[2][16 * 64];   //  4 KB

  const int tid  = threadIdx.x;
  const int lane = tid & 31;
  const int w    = tid >> 5;

  {
    uint4* d1 = reinterpret_cast<uint4*>(sW1);
    const uint4* s1 = reinterpret_cast<const uint4*>(w1sw);
    for (int i = tid; i < 128 * 64 / 8; i += 64) d1[i] = s1[i];
    uint4* d2 = reinterpret_cast<uint4*>(sW2);
    const uint4* s2 = reinterpret_cast<const uint4*>(w2sw);
    for (int i = tid; i < 64 * 64 / 8; i += 64) d2[i] = s2[i];
  }
  __syncthreads();

  const int gw = blockIdx.x * 2 + w;
  bf16* A1 = sA1[w];
  bf16* A2 = sA2[w];
  const int cn = lane & 15;
  const int rb = (lane >> 4) << 3;

  for (int it = 0; it < TPW; ++it) {
    const int tile = gw * TPW + it;
    if (tile >= NTILES) break;
    const int n0 = tile * 16;

    for (int idx = lane; idx < 16 * 16; idx += 32) {
      const int nr = idx >> 4, c4 = (idx & 15) << 2;
      float4 fa = *(const float4*)&nodes[(n0 + nr) * 64 + c4];
      float4 fb = *(const float4*)&agg[(n0 + nr) * 64 + c4];
      uint32_t* drow = reinterpret_cast<uint32_t*>(&A1[nr * 128]);
      *(uint2*)&drow[c4 >> 1] =
          make_uint2(pack2bf(fa.x, fa.y), pack2bf(fa.z, fa.w));
      *(uint2*)&drow[(64 + c4) >> 1] =
          make_uint2(pack2bf(fb.x, fb.y), pack2bf(fb.z, fb.w));
    }
    wave_lds_fence();

    v8f acc[4] = {};
#pragma unroll
    for (int ks = 0; ks < 4; ++ks) {
      v16bf a = load_a_frag(A1, 128, ks * 32, lane);
#pragma unroll
      for (int nb = 0; nb < 4; ++nb) {
        v16bf bm = load_b_frag(sW1, ks * 4 + nb, lane);
        acc[nb] = wmma_bf16(a, bm, acc[nb]);
      }
    }

#pragma unroll
    for (int nb = 0; nb < 4; ++nb) {
      const float bias = b1[nb * 16 + cn];
#pragma unroll
      for (int r = 0; r < 8; ++r) {
        float x = acc[nb][r] + bias;
        A2[(rb + r) * 64 + nb * 16 + cn] = (bf16)swish(x);
      }
    }
    wave_lds_fence();

    v8f acc2[4] = {};
#pragma unroll
    for (int ks = 0; ks < 2; ++ks) {
      v16bf a = load_a_frag(A2, 64, ks * 32, lane);
#pragma unroll
      for (int nb = 0; nb < 4; ++nb) {
        v16bf bm = load_b_frag(sW2, ks * 4 + nb, lane);
        acc2[nb] = wmma_bf16(a, bm, acc2[nb]);
      }
    }

#pragma unroll
    for (int nb = 0; nb < 4; ++nb) {
      const float bias = b2[nb * 16 + cn];
#pragma unroll
      for (int r = 0; r < 8; ++r) {
        out[(n0 + rb + r) * 64 + nb * 16 + cn] = acc2[nb][r] + bias;
      }
    }
    wave_lds_fence();
  }
}

// ---------------------------------------------------------------------------
// Launch. params flattened per jax pytree (sorted dict keys):
//   per layer: mW1, mW2, mb1, mb2, uW1, uW2, ub1, ub2 -> d_in[4 + 8*l + j]
// Workspace: [swizzled bf16 weights (128 KB pad)][agg 25.6 MB][mid 25.6 MB]
// ---------------------------------------------------------------------------
extern "C" void kernel_launch(void* const* d_in, const int* in_sizes, int n_in,
                              void* d_out, int out_size, void* d_ws,
                              size_t ws_size, hipStream_t stream) {
  const float* nodes0    = (const float*)d_in[0];
  const float* edges     = (const float*)d_in[1];
  const int*   senders   = (const int*)d_in[2];
  const int*   receivers = (const int*)d_in[3];

  bf16*  wsw = (bf16*)d_ws;
  float* agg = (float*)((char*)d_ws + (size_t)(1 << 17));
  float* mid = agg + (size_t)N_NODES * HID;

  const int PERL = 28672;  // bf16 elements of packed weights per layer

  for (int l = 0; l < 2; ++l) {
    const float* mW1 = (const float*)d_in[4 + 8 * l + 0];
    const float* mW2 = (const float*)d_in[4 + 8 * l + 1];
    const float* mb1 = (const float*)d_in[4 + 8 * l + 2];
    const float* mb2 = (const float*)d_in[4 + 8 * l + 3];
    const float* uW1 = (const float*)d_in[4 + 8 * l + 4];
    const float* uW2 = (const float*)d_in[4 + 8 * l + 5];
    const float* ub1 = (const float*)d_in[4 + 8 * l + 6];
    const float* ub2 = (const float*)d_in[4 + 8 * l + 7];

    bf16* mW1p = wsw + l * PERL;
    bf16* mW2p = mW1p + 12288;
    bf16* uW1p = mW1p + 16384;
    bf16* uW2p = mW1p + 24576;

    pack_w_kernel<<<(6144 + 255) / 256, 256, 0, stream>>>(
        mW1, (uint32_t*)mW1p, 6144);
    pack_w_kernel<<<(2048 + 255) / 256, 256, 0, stream>>>(
        mW2, (uint32_t*)mW2p, 2048);
    pack_w_kernel<<<(4096 + 255) / 256, 256, 0, stream>>>(
        uW1, (uint32_t*)uW1p, 4096);
    pack_w_kernel<<<(2048 + 255) / 256, 256, 0, stream>>>(
        uW2, (uint32_t*)uW2p, 2048);

    zero_kernel<<<(N_NODES * HID + 255) / 256, 256, 0, stream>>>(
        agg, N_NODES * HID);

    const float* nin  = (l == 0) ? nodes0 : mid;
    float*       nout = (l == 0) ? mid : (float*)d_out;

    const int eblocks = (ETILES + 2 * TPW - 1) / (2 * TPW);  // 3907
    const int nblocks = (NTILES + 2 * TPW - 1) / (2 * TPW);  // 391

    edge_kernel<<<eblocks, 64, 0, stream>>>(
        nin, edges, senders, receivers, mW1p, mW2p, mb1, mb2, agg);

    update_kernel<<<nblocks, 64, 0, stream>>>(
        nin, agg, uW1p, uW2p, ub1, ub2, nout);
  }
}
